// TopKSAE_80393197846570
// MI455X (gfx1250) — matrix operand order repaired and checked by reading, not compiled, not run on hardware
//
#include <hip/hip_runtime.h>

#define BATCH   32768
#define IN_DIM  768
#define HID     2048
#define TOPK    32

typedef __attribute__((ext_vector_type(16))) __bf16 v16bf;
typedef __attribute__((ext_vector_type(8)))  float  v8f;

__device__ __forceinline__ unsigned short f2bf(float f) {
  unsigned u = __float_as_uint(f);
  unsigned r = u + 0x7FFFu + ((u >> 16) & 1u);   // round-to-nearest-even
  return (unsigned short)(r >> 16);
}
__device__ __forceinline__ float bf2f(unsigned short h) {
  return __uint_as_float(((unsigned)h) << 16);
}

// ---------------------------------------------------------------------------
// Encoder: h_pre = x @ W_enc^T + b_enc, bf16 hi/lo split (3 WMMAs ~ fp32 acc)
// C-tile 128x256 per 256-thread block; wave w owns rows [16w,16w+16) x 256 cols
// ---------------------------------------------------------------------------
#define BM  128
#define BN  256
#define BK  32
#define LDA 48   // ushort stride (96B rows -> 32B-aligned frags, spread banks)
#define LDB 48

__global__ __launch_bounds__(256, 1) void sae_encoder(
    const float* __restrict__ x, const float* __restrict__ W_enc,
    const float* __restrict__ b_enc, float* __restrict__ h_pre)
{
  __shared__ unsigned short Ahi[BM * LDA];
  __shared__ unsigned short Alo[BM * LDA];
  __shared__ unsigned short Bhi[BN * LDB];
  __shared__ unsigned short Blo[BN * LDB];

  const int t    = threadIdx.x;
  const int lane = t & 31;
  const int wave = t >> 5;
  const int l16  = lane & 15;
  const int hi16 = lane >> 4;       // 0: lanes 0-15, 1: lanes 16-31
  const int m0   = blockIdx.x * BM;
  const int n0   = blockIdx.y * BN;

  v8f acc[16];
  {
    v8f z;
    for (int j = 0; j < 8; ++j) z[j] = 0.0f;
    for (int i = 0; i < 16; ++i) acc[i] = z;
  }

  const int arow = t >> 1;            // 0..127
  const int aseg = (t & 1) * 16;      // 0 or 16
  const float* aptr = x     + (size_t)(m0 + arow) * IN_DIM + aseg;
  const float* bptr = W_enc + (size_t)(n0 + t)    * IN_DIM;

  for (int k0 = 0; k0 < IN_DIM; k0 += BK) {
    __syncthreads();
    // ---- stage A (x tile 128x32) as bf16 hi/lo, identity K layout
    #pragma unroll
    for (int i = 0; i < 16; i += 2) {
      const int k  = aseg + i;
      const int g  = k >> 3;                           // group of 8 K values
      const int dk = k + ((g == 1) ? 0 : (g == 2) ? 0 : 0);
      const float f0 = aptr[k0 + i + 0];
      const float f1 = aptr[k0 + i + 1];
      const unsigned short h0 = f2bf(f0), h1 = f2bf(f1);
      const unsigned short q0 = f2bf(f0 - bf2f(h0)), q1 = f2bf(f1 - bf2f(h1));
      *(unsigned*)&Ahi[arow * LDA + dk] = (unsigned)h0 | ((unsigned)h1 << 16);
      *(unsigned*)&Alo[arow * LDA + dk] = (unsigned)q0 | ((unsigned)q1 << 16);
    }
    // ---- stage B (W_enc tile 256x32), identity K layout
    #pragma unroll
    for (int i = 0; i < 32; i += 2) {
      const float f0 = bptr[k0 + i + 0];
      const float f1 = bptr[k0 + i + 1];
      const unsigned short h0 = f2bf(f0), h1 = f2bf(f1);
      const unsigned short q0 = f2bf(f0 - bf2f(h0)), q1 = f2bf(f1 - bf2f(h1));
      *(unsigned*)&Bhi[t * LDB + i] = (unsigned)h0 | ((unsigned)h1 << 16);
      *(unsigned*)&Blo[t * LDB + i] = (unsigned)q0 | ((unsigned)q1 << 16);
    }
    if (k0 + BK < IN_DIM) {
      __builtin_prefetch(aptr + k0 + BK, 0, 1);   // global_prefetch next tiles
      __builtin_prefetch(bptr + k0 + BK, 0, 1);
    }
    __syncthreads();

    // ---- A fragments (single 32B LDS load each, contiguous k = 16*hi16 + s, same order as B)
    const int abase = (wave * 16 + l16) * LDA + hi16 * 16;
    const v16bf fa_hi = *(const v16bf*)&Ahi[abase];
    const v16bf fa_lo = *(const v16bf*)&Alo[abase];

    #pragma unroll
    for (int nt = 0; nt < 16; ++nt) {
      const int bbase = (nt * 16 + l16) * LDB + hi16 * 16;
      const v16bf fb_hi = *(const v16bf*)&Bhi[bbase];
      const v16bf fb_lo = *(const v16bf*)&Blo[bbase];
      acc[nt] = __builtin_amdgcn_wmma_f32_16x16x32_bf16(
          false, fa_hi, false, fb_hi, (short)0, acc[nt], false, false);
      acc[nt] = __builtin_amdgcn_wmma_f32_16x16x32_bf16(
          false, fa_lo, false, fb_hi, (short)0, acc[nt], false, false);
      acc[nt] = __builtin_amdgcn_wmma_f32_16x16x32_bf16(
          false, fa_hi, false, fb_lo, (short)0, acc[nt], false, false);
    }
  }

  // ---- epilogue: bias + store (C layout: VGPR g -> M = g + 8*hi16, N = l16)
  const int rbase = m0 + wave * 16 + hi16 * 8;
  #pragma unroll
  for (int nt = 0; nt < 16; ++nt) {
    const int col = n0 + nt * 16 + l16;
    const float bias = b_enc[col];
    #pragma unroll
    for (int g = 0; g < 8; ++g)
      h_pre[(size_t)(rbase + g) * HID + col] = acc[nt][g] + bias;
  }
}

// ---------------------------------------------------------------------------
// Top-K: one wave per row. Find 32nd-largest value T; h_sparse = (v>=T)?relu:0
// Also emit compact (idx,val)x32 per row for the sparse decoder.
// ---------------------------------------------------------------------------
__global__ __launch_bounds__(256, 1) void sae_topk(
    const float* __restrict__ h_pre, float* __restrict__ h_sparse,
    float* __restrict__ cval, int* __restrict__ cidx, int do_compact)
{
  __shared__ int cnt[8];
  const int lane = threadIdx.x & 31;
  const int wave = threadIdx.x >> 5;
  const int row  = blockIdx.x * 8 + wave;

  const float* src = h_pre + (size_t)row * HID;
  float v[64];
  #pragma unroll
  for (int i = 0; i < 16; ++i) {
    const float4 f = *(const float4*)(src + i * 128 + lane * 4);
    v[i * 4 + 0] = f.x; v[i * 4 + 1] = f.y;
    v[i * 4 + 2] = f.z; v[i * 4 + 3] = f.w;
  }

  if (do_compact) {
    if (lane == 0) cnt[wave] = 0;
    cval[(size_t)row * TOPK + lane] = 0.0f;   // pad defaults (LDS/global stores
    cidx[(size_t)row * TOPK + lane] = 0;      //  stay in order within a wave)
  }

  const float NEG = -__builtin_inff();
  float thr = __builtin_inff();
  #pragma unroll 1
  for (int j = 0; j < TOPK; ++j) {
    float local = NEG;
    #pragma unroll
    for (int i = 0; i < 64; ++i) {
      const float c = (v[i] < thr) ? v[i] : NEG;   // strict mask -> next max
      local = fmaxf(local, c);
    }
    #pragma unroll
    for (int off = 16; off > 0; off >>= 1)
      local = fmaxf(local, __shfl_xor(local, off, 32));
    thr = local;
  }

  float* dst = h_sparse + (size_t)row * HID;
  #pragma unroll
  for (int i = 0; i < 16; ++i) {
    float4 o;
    o.x = (v[i*4+0] >= thr) ? fmaxf(v[i*4+0], 0.0f) : 0.0f;
    o.y = (v[i*4+1] >= thr) ? fmaxf(v[i*4+1], 0.0f) : 0.0f;
    o.z = (v[i*4+2] >= thr) ? fmaxf(v[i*4+2], 0.0f) : 0.0f;
    o.w = (v[i*4+3] >= thr) ? fmaxf(v[i*4+3], 0.0f) : 0.0f;
    *(float4*)(dst + i * 128 + lane * 4) = o;
  }

  if (do_compact) {
    #pragma unroll 1
    for (int i = 0; i < 64; ++i) {
      if (v[i] >= thr) {
        const int p = atomicAdd(&cnt[wave], 1);   // ds_add_rtn_u32
        if (p < TOPK) {
          cval[(size_t)row * TOPK + p] = fmaxf(v[i], 0.0f);
          cidx[(size_t)row * TOPK + p] = (i >> 2) * 128 + lane * 4 + (i & 3);
        }
      }
    }
  }
}

// ---------------------------------------------------------------------------
// W_dec [768][2048] -> Wt [2048][768] (one-time 6.3MB, L2-resident afterwards)
// ---------------------------------------------------------------------------
__global__ __launch_bounds__(256, 1) void sae_transpose(
    const float* __restrict__ W_dec, float* __restrict__ Wt)
{
  const int id = blockIdx.x * 256 + threadIdx.x;   // 768*2048 total
  const int d = id >> 11;
  const int h = id & 2047;
  Wt[(size_t)h * IN_DIM + d] = W_dec[id];
}

// ---------------------------------------------------------------------------
// Decoder (sparse gather): recon[row] = b_dec + sum_j val[j] * Wcol(idx[j])
// 1 wave per row; 24 outputs/lane as 6 float4 accumulators.
// ---------------------------------------------------------------------------
__global__ __launch_bounds__(256, 1) void sae_decoder(
    const float* __restrict__ cval, const int* __restrict__ cidx,
    const float* __restrict__ W, const float* __restrict__ b_dec,
    float* __restrict__ recon, int w_transposed)
{
  const int lane = threadIdx.x & 31;
  const int wave = threadIdx.x >> 5;
  const int row  = blockIdx.x * 8 + wave;

  float4 a[6];
  #pragma unroll
  for (int c = 0; c < 6; ++c)
    a[c] = *(const float4*)(b_dec + c * 128 + lane * 4);

  #pragma unroll 1
  for (int j = 0; j < TOPK; ++j) {
    const float val = cval[(size_t)row * TOPK + j];
    const int   id  = cidx[(size_t)row * TOPK + j];
    if (w_transposed) {
      const float* wr = W + (size_t)id * IN_DIM;
      #pragma unroll
      for (int c = 0; c < 6; ++c) {
        const float4 w = *(const float4*)(wr + c * 128 + lane * 4);
        a[c].x += val * w.x; a[c].y += val * w.y;
        a[c].z += val * w.z; a[c].w += val * w.w;
      }
    } else {
      #pragma unroll
      for (int c = 0; c < 6; ++c) {
        const int d = c * 128 + lane * 4;
        a[c].x += val * W[(size_t)(d + 0) * HID + id];
        a[c].y += val * W[(size_t)(d + 1) * HID + id];
        a[c].z += val * W[(size_t)(d + 2) * HID + id];
        a[c].w += val * W[(size_t)(d + 3) * HID + id];
      }
    }
  }
  float* dst = recon + (size_t)row * IN_DIM;
  #pragma unroll
  for (int c = 0; c < 6; ++c)
    *(float4*)(dst + c * 128 + lane * 4) = a[c];
}

// Fallback decoder when scratch is too small for the compact list: scan dense
// h_sparse, broadcast nonzeros via ballot+shuffle.
__global__ __launch_bounds__(256, 1) void sae_decoder_dense(
    const float* __restrict__ h_sparse, const float* __restrict__ W_dec,
    const float* __restrict__ b_dec, float* __restrict__ recon)
{
  const int lane = threadIdx.x & 31;
  const int wave = threadIdx.x >> 5;
  const int row  = blockIdx.x * 8 + wave;

  float4 a[6];
  #pragma unroll
  for (int c = 0; c < 6; ++c)
    a[c] = *(const float4*)(b_dec + c * 128 + lane * 4);

  for (int h0 = 0; h0 < HID; h0 += 32) {
    const float val = h_sparse[(size_t)row * HID + h0 + lane];
    unsigned m = (unsigned)__ballot(val != 0.0f);
    while (m) {
      const int b = __ffs(m) - 1; m &= m - 1;
      const float bv = __shfl(val, b, 32);
      const int id = h0 + b;
      #pragma unroll
      for (int c = 0; c < 6; ++c) {
        const int d = c * 128 + lane * 4;
        a[c].x += bv * W_dec[(size_t)(d + 0) * HID + id];
        a[c].y += bv * W_dec[(size_t)(d + 1) * HID + id];
        a[c].z += bv * W_dec[(size_t)(d + 2) * HID + id];
        a[c].w += bv * W_dec[(size_t)(d + 3) * HID + id];
      }
    }
  }
  float* dst = recon + (size_t)row * IN_DIM;
  #pragma unroll
  for (int c = 0; c < 6; ++c)
    *(float4*)(dst + c * 128 + lane * 4) = a[c];
}

// ---------------------------------------------------------------------------
extern "C" void kernel_launch(void* const* d_in, const int* in_sizes, int n_in,
                              void* d_out, int out_size, void* d_ws, size_t ws_size,
                              hipStream_t stream)
{
  const float* x     = (const float*)d_in[0];
  const float* W_enc = (const float*)d_in[1];
  const float* b_enc = (const float*)d_in[2];
  const float* W_dec = (const float*)d_in[3];
  const float* b_dec = (const float*)d_in[4];

  float* out      = (float*)d_out;
  float* recon    = out;                                   // [32768 x 768]
  float* h_sparse = out + (size_t)BATCH * IN_DIM;          // [32768 x 2048]
  float* h_pre    = h_sparse + (size_t)BATCH * HID;        // [32768 x 2048]

  const size_t cval_bytes = (size_t)BATCH * TOPK * sizeof(float);
  const size_t cidx_bytes = (size_t)BATCH * TOPK * sizeof(int);
  const size_t wt_bytes   = (size_t)HID * IN_DIM * sizeof(float);
  const bool have_compact = ws_size >= cval_bytes + cidx_bytes;
  const bool have_wt      = ws_size >= cval_bytes + cidx_bytes + wt_bytes;

  float* cval = (float*)d_ws;
  int*   cidx = (int*)((char*)d_ws + cval_bytes);
  float* Wt   = (float*)((char*)d_ws + cval_bytes + cidx_bytes);

  if (have_wt)
    sae_transpose<<<dim3((IN_DIM * HID) / 256), dim3(256), 0, stream>>>(W_dec, Wt);

  sae_encoder<<<dim3(BATCH / BM, HID / BN), dim3(256), 0, stream>>>(
      x, W_enc, b_enc, h_pre);

  sae_topk<<<dim3(BATCH / 8), dim3(256), 0, stream>>>(
      h_pre, h_sparse, cval, cidx, have_compact ? 1 : 0);

  if (have_compact) {
    if (have_wt)
      sae_decoder<<<dim3(BATCH / 8), dim3(256), 0, stream>>>(
          cval, cidx, Wt, b_dec, recon, 1);
    else
      sae_decoder<<<dim3(BATCH / 8), dim3(256), 0, stream>>>(
          cval, cidx, W_dec, b_dec, recon, 0);
  } else {
    sae_decoder_dense<<<dim3(BATCH / 8), dim3(256), 0, stream>>>(
        h_sparse, W_dec, b_dec, recon);
  }
}
